// ChannelAttention_41016937677190
// MI455X (gfx1250) — matrix operand order repaired and checked by reading, not compiled, not run on hardware
//
#include <hip/hip_runtime.h>
#include <hip/hip_bf16.h>
#include <stdint.h>

// Problem constants (from reference: B=32, C=256, P=8192, fp32)
#define B_DIM 32
#define C_DIM 256
#define P_DIM 8192
#define KT    32      // K-chunk per WMMA (16x16x32 bf16)
#define KPAIR 16      // KT/2 packed bf16 pairs per row
#define KPAD  20      // padded row stride in dwords (multiple of 4 -> b128-aligned)

typedef __bf16 v16bf __attribute__((ext_vector_type(16)));
typedef float  v8f   __attribute__((ext_vector_type(8)));

union Frag16 { uint4 q[2]; v16bf v; };

__device__ __forceinline__ v8f v8f_zero() {
  v8f z;
#pragma unroll
  for (int j = 0; j < 8; ++j) z[j] = 0.f;
  return z;
}

// fp32 -> bf16 (round-half-up), packed pair via one v_perm_b32:
// result = { hi16(b+0x8000), hi16(a+0x8000) }  (a in [15:0], b in [31:16])
__device__ __forceinline__ uint32_t pack2_bf16(float a, float b) {
  uint32_t ua = __float_as_uint(a) + 0x8000u;
  uint32_t ub = __float_as_uint(b) + 0x8000u;
  // v_perm_b32: bytes 0-3 select from src1 (ua), 4-7 from src0 (ub)
  return __builtin_amdgcn_perm(ub, ua, 0x07060302u);
}

// A-matrix 16x32 bf16 fragment (ISA 7.12.2):
// lanes 0-15: kpairs {0..3, 8..11}; lanes 16-31: kpairs {4..7, 12..15}
__device__ __forceinline__ v16bf load_a_frag(const uint32_t* lds, int row, int half) {
  const uint32_t* r = lds + row * KPAD;
  Frag16 f;
  f.q[0] = *(const uint4*)(r + half * 4);
  f.q[1] = *(const uint4*)(r + 8 + half * 4);
  return f.v;
}

// B-matrix 32x16 bf16 fragment (same k order as A):
// lanes 0-15: kpairs {0..3, 8..11}; lanes 16-31: kpairs {4..7, 12..15}
__device__ __forceinline__ v16bf load_b_frag(const uint32_t* lds, int col, int half) {
  const uint32_t* r = lds + col * KPAD;
  Frag16 f;
  f.q[0] = *(const uint4*)(r + half * 4);
  f.q[1] = *(const uint4*)(r + 8 + half * 4);
  return f.v;
}

// ---------------------------------------------------------------------------
// Pass 1: energy = x[b] @ x[b]^T, fused inverted softmax.
// softmax(rowmax(e) - e) == exp(rowmin(e) - e) / sum  (numerically identical)
// grid = B*8 (32-row band per WG), block = 256 (8 waves, 2/SIMD).
// Wave w owns row-tile (w>>2) and column-quarter (w&3): 4 accum tiles.
// Pipeline: regs <- global(next) ; preload frags ; WMMA chain ; pack -> LDS(next).
// ---------------------------------------------------------------------------
__global__ __launch_bounds__(256) void pass1_energy_softmax(
    const float* __restrict__ x, float* __restrict__ attn) {
  const int b    = blockIdx.x >> 3;
  const int rblk = blockIdx.x & 7;
  const int tid  = threadIdx.x;
  const int wave = tid >> 5;
  const int lane = tid & 31;
  const int half = lane >> 4;
  const int ln16 = lane & 15;
  const int rt   = wave >> 2;   // row tile within band (0..1)
  const int cq   = wave & 3;    // column quarter: tiles [cq*4, cq*4+4)

  __shared__ __align__(16) uint32_t As[2][32 * KPAD];     // 2 x 2.5 KB
  __shared__ __align__(16) uint32_t Bs[2][C_DIM * KPAD];  // 2 x 20 KB
  __shared__ float RedMin[8][16];
  __shared__ float RedSum[8][16];

  const float* xb   = x + (size_t)b * C_DIM * P_DIM;
  const float* arow = xb + (size_t)(rblk * 32) * P_DIM;

  v8f acc[4];
#pragma unroll
  for (int t = 0; t < 4; ++t) acc[t] = v8f_zero();

  float2 rA[2];   // 32x16 pairs / 256 thr = 2
  float2 rB[16];  // 256x16 pairs / 256 thr = 16

  // prologue: issue chunk-0 global loads
#pragma unroll
  for (int j = 0; j < 2; ++j) {
    int idx = j * 256 + tid, r = idx >> 4, kp = idx & 15;
    rA[j] = *(const float2*)(arow + (size_t)r * P_DIM + kp * 2);
  }
#pragma unroll
  for (int j = 0; j < 16; ++j) {
    int idx = j * 256 + tid, r = idx >> 4, kp = idx & 15;
    rB[j] = *(const float2*)(xb + (size_t)r * P_DIM + kp * 2);
  }

  int buf = 0;
  for (int k0 = 0; k0 < P_DIM; k0 += KT) {
    // pack current chunk's registers into LDS[buf] (waits on loadcnt here)
#pragma unroll
    for (int j = 0; j < 2; ++j) {
      int idx = j * 256 + tid, r = idx >> 4, kp = idx & 15;
      As[buf][r * KPAD + kp] = pack2_bf16(rA[j].x, rA[j].y);
    }
#pragma unroll
    for (int j = 0; j < 16; ++j) {
      int idx = j * 256 + tid, r = idx >> 4, kp = idx & 15;
      Bs[buf][r * KPAD + kp] = pack2_bf16(rB[j].x, rB[j].y);
    }
    __syncthreads();

    // issue next chunk's global loads (in flight during WMMAs)
    const int kn = k0 + KT;
    if (kn < P_DIM) {
#pragma unroll
      for (int j = 0; j < 2; ++j) {
        int idx = j * 256 + tid, r = idx >> 4, kp = idx & 15;
        rA[j] = *(const float2*)(arow + (size_t)r * P_DIM + kn + kp * 2);
      }
#pragma unroll
      for (int j = 0; j < 16; ++j) {
        int idx = j * 256 + tid, r = idx >> 4, kp = idx & 15;
        rB[j] = *(const float2*)(xb + (size_t)r * P_DIM + kn + kp * 2);
      }
    }

    // preload all fragments, then issue back-to-back WMMAs
    v16bf a = load_a_frag(As[buf], rt * 16 + ln16, half);
    v16bf bb[4];
#pragma unroll
    for (int t = 0; t < 4; ++t)
      bb[t] = load_b_frag(Bs[buf], (cq * 4 + t) * 16 + ln16, half);
#pragma unroll
    for (int t = 0; t < 4; ++t)
      acc[t] = __builtin_amdgcn_wmma_f32_16x16x32_bf16(
          false, a, false, bb[t], (short)0, acc[t], false, false);
    buf ^= 1;
  }

  // ---- fused softmax over full 256-column rows (cross-wave via LDS) ----
  // D-tile layout: VGPR v <-> row16 (v + 8*half), lane%16 <-> column in tile.
#pragma unroll
  for (int v = 0; v < 8; ++v) {
    float m = acc[0][v];
#pragma unroll
    for (int t = 1; t < 4; ++t) m = fminf(m, acc[t][v]);
#pragma unroll
    for (int off = 1; off < 16; off <<= 1)
      m = fminf(m, __shfl_xor(m, off, 16));
    if (ln16 == 0) RedMin[wave][v + 8 * half] = m;
  }
  __syncthreads();
  float rmin[8];
#pragma unroll
  for (int v = 0; v < 8; ++v) {
    int r = v + 8 * half;
    float m = RedMin[rt * 4 + 0][r];
#pragma unroll
    for (int q = 1; q < 4; ++q) m = fminf(m, RedMin[rt * 4 + q][r]);
    rmin[v] = m;
  }
#pragma unroll
  for (int v = 0; v < 8; ++v) {
    float s = 0.f;
#pragma unroll
    for (int t = 0; t < 4; ++t) {
      float e = __expf(rmin[v] - acc[t][v]);
      acc[t][v] = e;
      s += e;
    }
#pragma unroll
    for (int off = 1; off < 16; off <<= 1)
      s += __shfl_xor(s, off, 16);
    if (ln16 == 0) RedSum[wave][v + 8 * half] = s;
  }
  __syncthreads();

  float* ao = attn + (size_t)b * C_DIM * C_DIM;
#pragma unroll
  for (int v = 0; v < 8; ++v) {
    int r = v + 8 * half;
    float s = RedSum[rt * 4 + 0][r];
#pragma unroll
    for (int q = 1; q < 4; ++q) s += RedSum[rt * 4 + q][r];
    float rinv = 1.f / s;
    int row = rblk * 32 + rt * 16 + r;
#pragma unroll
    for (int t = 0; t < 4; ++t) {
      ao[(size_t)row * C_DIM + (cq * 4 + t) * 16 + ln16] = acc[t][v] * rinv;
    }
  }
}

// ---------------------------------------------------------------------------
// Pass 2: out = alpha * (attn[b] @ x[b]) + x[b].
// grid = B * 2 * 64 (128-row half x 128-col block), block = 256 (8 waves).
// Wave w owns row-tile w (16 rows) across all 8 column tiles: 8 accum tiles.
// Same register->LDS software pipeline, K = 256 (8 chunks).
// ---------------------------------------------------------------------------
__global__ __launch_bounds__(256) void pass2_apply(
    const float* __restrict__ x, const float* __restrict__ attn,
    const float* __restrict__ alpha, float* __restrict__ out) {
  const int b    = blockIdx.x >> 7;
  const int rh   = (blockIdx.x >> 6) & 1;  // row half (128 rows)
  const int pblk = blockIdx.x & 63;
  const int p0   = pblk * 128;
  const int tid  = threadIdx.x;
  const int wave = tid >> 5;
  const int lane = tid & 31;
  const int half = lane >> 4;
  const int ln16 = lane & 15;

  __shared__ __align__(16) uint32_t As[2][128 * KPAD];  // attn, 2 x 10 KB
  __shared__ __align__(16) uint32_t Bs[2][128 * KPAD];  // x slab (transposed pack)

  const float* ab = attn + (size_t)b * C_DIM * C_DIM + (size_t)(rh * 128) * C_DIM;
  const float* xb = x + (size_t)b * C_DIM * P_DIM;

  v8f acc[8];
#pragma unroll
  for (int t = 0; t < 8; ++t) acc[t] = v8f_zero();

  float2 rA[8];              // 128x16 pairs / 256 thr = 8
  float  rB0[8], rB1[8];     // 128x16 pairs / 256 thr = 8 (two strided loads each)

  // prologue: chunk 0
#pragma unroll
  for (int j = 0; j < 8; ++j) {
    int idx = j * 256 + tid, r = idx >> 4, kp = idx & 15;
    rA[j] = *(const float2*)(ab + (size_t)r * C_DIM + kp * 2);
  }
#pragma unroll
  for (int j = 0; j < 8; ++j) {
    int idx = j * 256 + tid, p = idx & 127, kp = idx >> 7;
    const float* s0 = xb + (size_t)(2 * kp) * P_DIM + p0 + p;
    rB0[j] = s0[0];
    rB1[j] = s0[P_DIM];
  }

  int buf = 0;
  for (int k0 = 0; k0 < C_DIM; k0 += KT) {
#pragma unroll
    for (int j = 0; j < 8; ++j) {
      int idx = j * 256 + tid, r = idx >> 4, kp = idx & 15;
      As[buf][r * KPAD + kp] = pack2_bf16(rA[j].x, rA[j].y);
    }
#pragma unroll
    for (int j = 0; j < 8; ++j) {
      int idx = j * 256 + tid, p = idx & 127, kp = idx >> 7;
      Bs[buf][p * KPAD + kp] = pack2_bf16(rB0[j], rB1[j]);
    }
    __syncthreads();

    const int kn = k0 + KT;
    if (kn < C_DIM) {
#pragma unroll
      for (int j = 0; j < 8; ++j) {
        int idx = j * 256 + tid, r = idx >> 4, kp = idx & 15;
        rA[j] = *(const float2*)(ab + (size_t)r * C_DIM + kn + kp * 2);
      }
#pragma unroll
      for (int j = 0; j < 8; ++j) {
        int idx = j * 256 + tid, p = idx & 127, kp = idx >> 7;
        const float* s0 = xb + (size_t)(kn + 2 * kp) * P_DIM + p0 + p;
        rB0[j] = s0[0];
        rB1[j] = s0[P_DIM];
      }
    }

    // preload all fragments, then issue back-to-back WMMAs
    v16bf a = load_a_frag(As[buf], wave * 16 + ln16, half);
    v16bf bb[8];
#pragma unroll
    for (int ct = 0; ct < 8; ++ct)
      bb[ct] = load_b_frag(Bs[buf], ct * 16 + ln16, half);
#pragma unroll
    for (int ct = 0; ct < 8; ++ct)
      acc[ct] = __builtin_amdgcn_wmma_f32_16x16x32_bf16(
          false, a, false, bb[ct], (short)0, acc[ct], false, false);
    buf ^= 1;
  }

  const float al = alpha[0];
#pragma unroll
  for (int v = 0; v < 8; ++v) {
    int row = rh * 128 + wave * 16 + v + 8 * half;
#pragma unroll
    for (int ct = 0; ct < 8; ++ct) {
      size_t idx = ((size_t)b * C_DIM + row) * P_DIM + p0 + ct * 16 + ln16;
      out[idx] = fmaf(al, acc[ct][v], x[idx]);
    }
  }
}

extern "C" void kernel_launch(void* const* d_in, const int* in_sizes, int n_in,
                              void* d_out, int out_size, void* d_ws, size_t ws_size,
                              hipStream_t stream) {
  (void)in_sizes; (void)n_in; (void)out_size; (void)ws_size;
  const float* x     = (const float*)d_in[0];
  const float* alpha = (const float*)d_in[1];
  float* out  = (float*)d_out;
  float* attn = (float*)d_ws;  // B*C*C fp32 = 8 MB scratch

  pass1_energy_softmax<<<dim3(B_DIM * 8), dim3(256), 0, stream>>>(x, attn);
  pass2_apply<<<dim3(B_DIM * 2 * 64), dim3(256), 0, stream>>>(x, attn, alpha, out);
}
